// fromScratchLSTM_41729902248142
// MI455X (gfx1250) — matrix operand order repaired and checked by reading, not compiled, not run on hardware
//
#include <hip/hip_runtime.h>

// ---------------------------------------------------------------------------
// LSTM (B=64, S=512, I=256, H=512) for MI455X / gfx1250, wave32 + WMMA bf16.
//
//  * Per step s: gates = [x_s | h] @ [[Wi];[Wh]] + b  -> fused K=768 GEMM,
//    M=64, N=2048 (4 gates x 512), v_wmma_f32_16x16x32_bf16 (f32 accum).
//  * Persistent kernel: 16 blocks x 256 threads = 128 waves, one wave per
//    (16-row m-tile, 16-col h-tile) owning all 4 gate accumulators; cell
//    state c stays in registers across all 512 steps.
//  * h double-buffered in bf16 in workspace; monotonic atomic grid barrier.
//  * Weights pre-swizzled into B-fragment lanes, contiguous k = 16*(lane/16)+e (same as A);
//    wave-uniform (readfirstlane) gate offsets are laundered through inline
//    asm ("+s") each step so the 96 fragment loads per step stream from L2
//    instead of being hoisted out of the time loop and spilled to scratch.
//  * Activations use v_exp_f32 + v_rcp_f32 (no IEEE divide sequences).
// ---------------------------------------------------------------------------

typedef __attribute__((ext_vector_type(16))) __bf16 v16bf;
typedef __attribute__((ext_vector_type(8)))  float  v8f;

namespace {
constexpr int B  = 64;
constexpr int S  = 512;
constexpr int I  = 256;
constexpr int H  = 512;
constexpr int K  = I + H;        // 768 fused K dimension
constexpr int KB = K / 32;       // 24 K-blocks of 32
constexpr int NWG = 16;          // persistent workgroups (4 m-tiles x 4 h-groups)
}

// ---------------- init kernels -------------------------------------------

__global__ void cvt_x_bf16(const float* __restrict__ x, __bf16* __restrict__ o, int n) {
    int i = blockIdx.x * blockDim.x + threadIdx.x;
    if (i < n) o[i] = (__bf16)x[i];
}

// Pack [Wi;Wh] (per gate) into WMMA B-fragment order:
//   pw[(((g*32+ht)*KB+kb)*32+lane)*16+e] = Wcat[g][kb*32 + (lane&16?16:0) + e][ht*16 + (lane&15)]
__global__ void pack_weights(const float* __restrict__ Wii, const float* __restrict__ Whi,
                             const float* __restrict__ Wif, const float* __restrict__ Whf,
                             const float* __restrict__ Wig, const float* __restrict__ Whg,
                             const float* __restrict__ Wio, const float* __restrict__ Who,
                             __bf16* __restrict__ pw) {
    unsigned t = blockIdx.x * 256u + threadIdx.x;
    unsigned e    = t & 15u;  unsigned r = t >> 4;
    unsigned lane = r & 31u;  r >>= 5;
    unsigned kb   = r % 24u;  r /= 24u;
    unsigned ht   = r & 31u;  unsigned g = r >> 5;

    unsigned k    = kb * 32u + ((lane & 16u) ? 16u : 0u) + e;
    unsigned colh = ht * 16u + (lane & 15u);

    const float* Wi; const float* Wh;
    switch (g) {
        case 0:  Wi = Wii; Wh = Whi; break;
        case 1:  Wi = Wif; Wh = Whf; break;
        case 2:  Wi = Wig; Wh = Whg; break;
        default: Wi = Wio; Wh = Who; break;
    }
    float v = (k < (unsigned)I) ? Wi[(size_t)k * H + colh]
                                : Wh[(size_t)(k - I) * H + colh];
    pw[t] = (__bf16)v;
}

// ---------------- persistent LSTM kernel ---------------------------------

// sigmoid(x) = rcp(1 + exp(-x)) : v_exp_f32 + v_rcp_f32, no IEEE divide.
__device__ __forceinline__ float fast_sigmoid(float x) {
    return __builtin_amdgcn_rcpf(1.0f + __expf(-x));
}
// tanh(x) = (e-1)*rcp(e+1), e = exp(2x), clamped to avoid inf*rcp(inf).
__device__ __forceinline__ float fast_tanh(float x) {
    x = fminf(fmaxf(x, -15.0f), 15.0f);
    float e = __expf(2.0f * x);
    return (e - 1.0f) * __builtin_amdgcn_rcpf(e + 1.0f);
}

__global__ __launch_bounds__(256) void lstm_persistent(
        const __bf16* __restrict__ xbf,     // [B,S,I] bf16
        const __bf16* __restrict__ pw,      // packed weight fragments
        __bf16* __restrict__ h0buf,         // [B,H] bf16 ping
        __bf16* __restrict__ h1buf,         // [B,H] bf16 pong
        unsigned* __restrict__ bar,         // grid barrier counter (init 0)
        const float* __restrict__ bi,  const float* __restrict__ bfv,
        const float* __restrict__ bg,  const float* __restrict__ bo,
        float* __restrict__ out)            // hs [B,S,H] ++ hT [B,H] ++ cT [B,H]
{
    __shared__ __bf16 Asm[16 * K];          // 24 KB: [x_s | h] tile, 16 rows x 768

    const int tid   = threadIdx.x;
    const int lane  = tid & 31;
    const int wave  = tid >> 5;
    const int mtile = blockIdx.x & 3;                  // batch tile (16 rows)
    const int ht    = ((blockIdx.x >> 2) * 8) + wave;  // h tile 0..31 (16 cols)

    const int col   = ht * 16 + (lane & 15);     // output h column of this lane
    const int asel  = lane >> 4;                 // A-fragment half select
    const int arow  = lane & 15;                 // A-fragment source row
    const int rbase = mtile * 16 + (asel << 3);  // C/D row base (+r, r=0..7)

    const float bii = bi[col],  bff = bfv[col];
    const float bgg = bg[col],  boo = bo[col];

    // Wave-uniform per-gate fragment BYTE offsets into pw. ht is uniform
    // across the wave, so readfirstlane is exact; it also moves the value
    // into an SGPR so the "+s" laundering below is legal.
    const int u0 = __builtin_amdgcn_readfirstlane((int)((((0 * 32 + ht) * KB) * 512) * 2));
    const int u1 = __builtin_amdgcn_readfirstlane((int)((((1 * 32 + ht) * KB) * 512) * 2));
    const int u2 = __builtin_amdgcn_readfirstlane((int)((((2 * 32 + ht) * KB) * 512) * 2));
    const int u3 = __builtin_amdgcn_readfirstlane((int)((((3 * 32 + ht) * KB) * 512) * 2));
    const int lofs = lane * 16;                  // element offset within fragment

    float c[8];
    #pragma unroll
    for (int r = 0; r < 8; r++) c[r] = 0.0f;

    __bf16* hb[2] = { h0buf, h1buf };
    const size_t hsSize = (size_t)B * S * H;

    for (int s = 0; s < S; s++) {
        const __bf16* hprev = hb[s & 1];

        // -------- cooperative fill of A = [x_s | h] (16 x 768 bf16) --------
        // 16 threads per row, b128 granularity: 32 x-quads + 64 h-quads/row.
        {
            const int r = tid >> 4;        // row 0..15
            const int q = tid & 15;        // quad phase
            const uint4* xsrc = (const uint4*)(xbf + ((size_t)(mtile * 16 + r) * S + s) * I);
            const uint4* hsrc = (const uint4*)(hprev + (size_t)(mtile * 16 + r) * H);
            uint4* dst = (uint4*)(Asm + r * K);
            #pragma unroll
            for (int j = 0; j < 2; j++) dst[q + 16 * j]      = xsrc[q + 16 * j];
            #pragma unroll
            for (int j = 0; j < 4; j++) dst[32 + q + 16 * j] = hsrc[q + 16 * j];
        }
        __syncthreads();

        // Launder the (step-invariant) uniform gate offsets so the 96
        // fragment loads cannot be hoisted out of the time loop.
        int o0 = u0, o1 = u1, o2 = u2, o3 = u3;
        asm volatile("" : "+s"(o0), "+s"(o1), "+s"(o2), "+s"(o3));
        const __bf16* wq0 = (const __bf16*)((const char*)pw + o0);
        const __bf16* wq1 = (const __bf16*)((const char*)pw + o1);
        const __bf16* wq2 = (const __bf16*)((const char*)pw + o2);
        const __bf16* wq3 = (const __bf16*)((const char*)pw + o3);

        // -------- K loop: 24 blocks x 4 gates of wmma bf16 -----------------
        v8f acc_i, acc_f, acc_g, acc_o;
        #pragma unroll
        for (int r = 0; r < 8; r++) {
            acc_i[r] = bii; acc_f[r] = bff; acc_g[r] = bgg; acc_o[r] = boo;
        }

        const uint4* abase = (const uint4*)(Asm + arow * K);
        #pragma unroll 2
        for (int kb = 0; kb < KB; kb++) {
            union { uint4 q[2]; v16bf v; } A;
            A.q[0] = abase[kb * 4 + 2 * asel];      // K dwords {0..3} / {8..11}
            A.q[1] = abase[kb * 4 + 2 * asel + 1];  // K dwords {4..7} / {12..15}

            const int fo = kb * 512 + lofs;
            v16bf b0 = *(const v16bf*)(wq0 + fo);
            v16bf b1 = *(const v16bf*)(wq1 + fo);
            v16bf b2 = *(const v16bf*)(wq2 + fo);
            v16bf b3 = *(const v16bf*)(wq3 + fo);

            acc_i = __builtin_amdgcn_wmma_f32_16x16x32_bf16(false, A.v, false, b0, (short)0, acc_i, false, false);
            acc_f = __builtin_amdgcn_wmma_f32_16x16x32_bf16(false, A.v, false, b1, (short)0, acc_f, false, false);
            acc_g = __builtin_amdgcn_wmma_f32_16x16x32_bf16(false, A.v, false, b2, (short)0, acc_g, false, false);
            acc_o = __builtin_amdgcn_wmma_f32_16x16x32_bf16(false, A.v, false, b3, (short)0, acc_o, false, false);
        }

        // -------- cell update + stores -------------------------------------
        __bf16* hnext = hb[(s + 1) & 1];
        #pragma unroll
        for (int r = 0; r < 8; r++) {
            float iv = fast_sigmoid(acc_i[r]);
            float fv = fast_sigmoid(acc_f[r]);
            float gv = fast_tanh(acc_g[r]);
            float ov = fast_sigmoid(acc_o[r]);
            c[r] = fv * c[r] + iv * gv;
            float hv = ov * fast_tanh(c[r]);

            int b = rbase + r;
            out[((size_t)b * S + s) * H + col] = hv;
            hnext[(size_t)b * H + col] = (__bf16)hv;
            if (s == S - 1) {
                out[hsSize + (size_t)b * H + col]                 = hv;   // hT
                out[hsSize + (size_t)B * H + (size_t)b * H + col] = c[r]; // cT
            }
        }

        // -------- grid-wide hand-off of h ----------------------------------
        if (s < S - 1) {
            __threadfence();
            __syncthreads();
            if (tid == 0) {
                __hip_atomic_fetch_add(bar, 1u, __ATOMIC_RELEASE, __HIP_MEMORY_SCOPE_AGENT);
                const unsigned target = (unsigned)(s + 1) * NWG;
                while (__hip_atomic_load(bar, __ATOMIC_ACQUIRE, __HIP_MEMORY_SCOPE_AGENT) < target) {
                    __builtin_amdgcn_s_sleep(1);
                }
            }
            __syncthreads();
        }
    }
}

// ---------------- host side ----------------------------------------------

extern "C" void kernel_launch(void* const* d_in, const int* in_sizes, int n_in,
                              void* d_out, int out_size, void* d_ws, size_t ws_size,
                              hipStream_t stream) {
    (void)in_sizes; (void)n_in; (void)out_size; (void)ws_size;

    const float* x   = (const float*)d_in[0];
    const float* Wii = (const float*)d_in[1];
    const float* Whi = (const float*)d_in[2];
    const float* bi  = (const float*)d_in[3];
    const float* Wif = (const float*)d_in[4];
    const float* Whf = (const float*)d_in[5];
    const float* bfv = (const float*)d_in[6];
    const float* Wig = (const float*)d_in[7];
    const float* Whg = (const float*)d_in[8];
    const float* bg  = (const float*)d_in[9];
    const float* Wio = (const float*)d_in[10];
    const float* Who = (const float*)d_in[11];
    const float* bo  = (const float*)d_in[12];
    float* out = (float*)d_out;

    // workspace layout (all offsets 256B aligned)
    char* ws = (char*)d_ws;
    const size_t xbf_bytes = (size_t)B * S * I * 2;              // 16 MB
    const size_t pw_bytes  = (size_t)4 * 32 * KB * 32 * 16 * 2;  // 3 MB
    const size_t h_bytes   = (size_t)B * H * 2;                  // 64 KB each

    __bf16* xbf = (__bf16*)ws;
    __bf16* pw  = (__bf16*)(ws + xbf_bytes);
    __bf16* h0  = (__bf16*)(ws + xbf_bytes + pw_bytes);
    __bf16* h1  = (__bf16*)(ws + xbf_bytes + pw_bytes + h_bytes);

    // 1) x -> bf16
    const int nx = B * S * I;
    cvt_x_bf16<<<(nx + 255) / 256, 256, 0, stream>>>(x, xbf, nx);

    // 2) pre-swizzle weights into B-fragment layout
    const int npw = 4 * 32 * KB * 32 * 16;
    pack_weights<<<npw / 256, 256, 0, stream>>>(Wii, Whi, Wif, Whf, Wig, Whg, Wio, Who, pw);

    // 3) zero h0/h1 and the barrier counter (contiguous region)
    unsigned* bar = (unsigned*)(ws + xbf_bytes + pw_bytes + 2 * h_bytes);
    (void)hipMemsetAsync(h0, 0, 2 * h_bytes + 256, stream);

    // 4) persistent recurrent kernel: 16 WGs x 256 threads (128 waves)
    lstm_persistent<<<NWG, 256, 0, stream>>>(xbf, pw, h0, h1, bar,
                                             bi, bfv, bg, bo, out);
}
